// metricloss_83528523972733
// MI455X (gfx1250) — compile-verified, matrix operand order checked by reading, not run on hardware
//
#include <hip/hip_runtime.h>
#include <hip/hip_bf16.h>

typedef __attribute__((ext_vector_type(16))) _Float16 v16h;
typedef __attribute__((ext_vector_type(8)))  _Float16 v8h;
typedef __attribute__((ext_vector_type(4)))  _Float16 v4h;
typedef __attribute__((ext_vector_type(8)))  float    v8f;

#define INV_TEMP 10.0f
#define DK 128            // embedding dim (fixed by reference setup)
#define MAXN 8192

// xor-shuffle via immediate ds_swizzle (group-of-32: {xor<<10 | or<<5 | and})
template <int MASK>
__device__ __forceinline__ float xsh(float x) {
    return __builtin_bit_cast(float,
        __builtin_amdgcn_ds_swizzle(__builtin_bit_cast(int, x),
                                    (MASK << 10) | 0x1f));
}

// ---------------------------------------------------------------- zeroing ---
__global__ void metricloss_zero_kernel(float* __restrict__ rowsum,
                                       float* __restrict__ out, int n) {
    int i = blockIdx.x * blockDim.x + threadIdx.x;
    if (i < n) rowsum[i] = 0.0f;
    if (i == 0) out[0] = 0.0f;
}

// ------------------------------------------------- row-normalize to f16 -----
// one wave32 per row of 128 floats; each lane owns 4 contiguous floats
__global__ void metricloss_normalize_kernel(const float* __restrict__ X,
                                            _Float16* __restrict__ Xh, int N) {
    int row  = (int)((blockIdx.x * blockDim.x + threadIdx.x) >> 5);
    int lane = threadIdx.x & 31;
    if (row >= N) return;
    const float4 v = *(const float4*)(X + (size_t)row * DK + lane * 4);
    float ss = v.x * v.x + v.y * v.y + v.z * v.z + v.w * v.w;
    ss += xsh<16>(ss);
    ss += xsh<8>(ss);
    ss += xsh<4>(ss);
    ss += xsh<2>(ss);
    ss += xsh<1>(ss);
    float scale = 1.0f / fmaxf(sqrtf(ss), 1e-12f);
    v4h h;
    h[0] = (_Float16)(v.x * scale);
    h[1] = (_Float16)(v.y * scale);
    h[2] = (_Float16)(v.z * scale);
    h[3] = (_Float16)(v.w * scale);
    *(v4h*)(Xh + (size_t)row * DK + lane * 4) = h;
}

// ------------------------------------------------ fragment load (16-bit) ----
// ISA 16-bit A layout, K-chunk of 32: lane L<16 holds K {o..o+7, o+16..o+23}
// with o = kk; lanes 16-31 same with o = kk+8.  Two 16B contiguous loads.
__device__ __forceinline__ v16h metricloss_load_frag(
        const _Float16* __restrict__ row, int o) {
    v8h lo = *(const v8h*)(row + o);
    v8h hi = *(const v8h*)(row + o + 16);
    return __builtin_shufflevector(lo, hi,
        0, 1, 2, 3, 4, 5, 6, 7, 8, 9, 10, 11, 12, 13, 14, 15);
}

// --------------------- fused GEMM strip (16x64) + exp + mask + row-sums -----
// one wave per 16x64 strip: A fragments (4 K-steps) kept in registers and
// reused across 4 B sub-tiles -> 16 v_wmma per wave, one row-reduction.
__global__ void __launch_bounds__(256)
metricloss_simtile_kernel(const _Float16* __restrict__ Xh,
                          const int* __restrict__ label,
                          float* __restrict__ rowsum, int T) {
    const int stripsPerRow = T >> 2;
    int wg = blockIdx.x * 8 + (int)(threadIdx.x >> 5);   // wave-uniform
    if (wg >= T * stripsPerRow) return;
    int ti  = wg / stripsPerRow;
    int tjg = wg - ti * stripsPerRow;
    if (tjg < (ti >> 2)) return;                         // below-triangle strip

    int lane = threadIdx.x & 31;
    int hw   = lane >> 4;        // half-wave id (0/1)
    int l15  = lane & 15;
    int i0   = ti << 4;
    int ibase = i0 + (hw << 3);

    const _Float16* arow = Xh + (size_t)(i0 + l15) * DK;

    // A fragments for all 4 K-steps, kept live across the strip
    v16h a0 = metricloss_load_frag(arow, 0   + (hw << 3));
    v16h a1 = metricloss_load_frag(arow, 32  + (hw << 3));
    v16h a2 = metricloss_load_frag(arow, 64  + (hw << 3));
    v16h a3 = metricloss_load_frag(arow, 96  + (hw << 3));

    // labels for this strip's 8 rows (uniform across half-wave -> L1 broadcast)
    int labI[8];
#pragma unroll
    for (int r = 0; r < 8; ++r) labI[r] = label[ibase + r];

    float rowacc[8] = {0.f, 0.f, 0.f, 0.f, 0.f, 0.f, 0.f, 0.f};

#pragma unroll
    for (int s = 0; s < 4; ++s) {
        int tj = (tjg << 2) + s;
        if (tj >= ti) {                                  // wave-uniform
            int j0 = tj << 4;
            const _Float16* brow = Xh + (size_t)(j0 + l15) * DK;
            if (s < 3)                                   // hint next B sub-tile
                __builtin_prefetch(brow + 16 * DK, 0, 3);

            v8f c = {};
            c = __builtin_amdgcn_wmma_f32_16x16x32_f16(false, a0, false,
                    metricloss_load_frag(brow, 0   + (hw << 3)),
                    (short)0, c, false, false);
            c = __builtin_amdgcn_wmma_f32_16x16x32_f16(false, a1, false,
                    metricloss_load_frag(brow, 32  + (hw << 3)),
                    (short)0, c, false, false);
            c = __builtin_amdgcn_wmma_f32_16x16x32_f16(false, a2, false,
                    metricloss_load_frag(brow, 64  + (hw << 3)),
                    (short)0, c, false, false);
            c = __builtin_amdgcn_wmma_f32_16x16x32_f16(false, a3, false,
                    metricloss_load_frag(brow, 96  + (hw << 3)),
                    (short)0, c, false, false);

            // C layout: c[r] -> row (ibase + r), col (j0 + l15)
            int gj   = j0 + l15;
            int labj = label[gj];
            float colsum = 0.0f;
#pragma unroll
            for (int r = 0; r < 8; ++r) {
                int gi = ibase + r;
                float v = 0.0f;
                if (labI[r] == labj && gi != gj)
                    v = __expf(c[r] * INV_TEMP);
                rowacc[r] += v;                          // in-lane row partial
                colsum    += v;                          // in-lane col partial
            }
            if (tj != ti) {                              // symmetric doubling
                colsum += xsh<16>(colsum);               // combine half-waves
                if (hw == 0) atomicAdd(&rowsum[gj], colsum);
            }
        }
    }

    // ---- value-halving butterfly: 8 row sums across 16 lanes in 8 swizzles
    // stage xor 8: keep 4 rows, give 4
    float w0, w1, w2, w3;
    {
        bool up = (l15 & 8) != 0;                        // set: keep rows 4-7
        float g0 = up ? rowacc[0] : rowacc[4];
        float g1 = up ? rowacc[1] : rowacc[5];
        float g2 = up ? rowacc[2] : rowacc[6];
        float g3 = up ? rowacc[3] : rowacc[7];
        w0 = (up ? rowacc[4] : rowacc[0]) + xsh<8>(g0);
        w1 = (up ? rowacc[5] : rowacc[1]) + xsh<8>(g1);
        w2 = (up ? rowacc[6] : rowacc[2]) + xsh<8>(g2);
        w3 = (up ? rowacc[7] : rowacc[3]) + xsh<8>(g3);
    }
    // stage xor 4: keep 2
    float u0, u1;
    {
        bool up = (l15 & 4) != 0;
        float g0 = up ? w0 : w2;
        float g1 = up ? w1 : w3;
        u0 = (up ? w2 : w0) + xsh<4>(g0);
        u1 = (up ? w3 : w1) + xsh<4>(g1);
    }
    // stage xor 2: keep 1
    float t0;
    {
        bool up = (l15 & 2) != 0;
        float g = up ? u0 : u1;
        t0 = (up ? u1 : u0) + xsh<2>(g);
    }
    // stage xor 1: pairs hold the same row
    t0 += xsh<1>(t0);
    // lane l holds total for row ibase + ((l>>1)&7)
    if ((l15 & 1) == 0)
        atomicAdd(&rowsum[ibase + ((l15 >> 1) & 7)], t0);
}

// ---------------------------------------------- n_i, divide, global reduce --
__global__ void metricloss_finalize_kernel(const int* __restrict__ label,
                                           const float* __restrict__ rowsum,
                                           float* __restrict__ out, int N) {
    __shared__ int slab[MAXN];
    for (int t = threadIdx.x; t < N; t += blockDim.x) slab[t] = label[t];
    __syncthreads();

    int i = blockIdx.x * blockDim.x + threadIdx.x;
    float contrib = 0.0f;
    if (i < N) {
        int li = slab[i];
        int cnt = 0;
        for (int j = 0; j < N; j += 4) {
            cnt += (slab[j]     == li) ? 1 : 0;
            cnt += (slab[j + 1] == li) ? 1 : 0;
            cnt += (slab[j + 2] == li) ? 1 : 0;
            cnt += (slab[j + 3] == li) ? 1 : 0;
        }
        int ni = cnt - 1;
        if (ni > 0) contrib = rowsum[i] / (float)ni;
    }
    contrib += xsh<16>(contrib);
    contrib += xsh<8>(contrib);
    contrib += xsh<4>(contrib);
    contrib += xsh<2>(contrib);
    contrib += xsh<1>(contrib);
    if ((threadIdx.x & 31) == 0) atomicAdd(out, contrib);
}

// ---------------------------------------------------------------- launch ----
extern "C" void kernel_launch(void* const* d_in, const int* in_sizes, int n_in,
                              void* d_out, int out_size, void* d_ws, size_t ws_size,
                              hipStream_t stream) {
    const float* X     = (const float*)d_in[0];   // [N, 128] fp32
    const int*   label = (const int*)d_in[1];     // [N] int32
    float*       out   = (float*)d_out;           // scalar fp32

    int N = in_sizes[1];                          // 8192
    // workspace: [ Xh f16 N*128 | rowsum f32 N ]
    _Float16* Xh     = (_Float16*)d_ws;
    float*    rowsum = (float*)((char*)d_ws + (size_t)N * DK * sizeof(_Float16));

    metricloss_zero_kernel<<<(N + 255) / 256, 256, 0, stream>>>(rowsum, out, N);

    // one wave per row -> N*32 threads
    metricloss_normalize_kernel<<<(N * 32 + 255) / 256, 256, 0, stream>>>(X, Xh, N);

    int T = N / 16;                               // 512 tiles per dim
    int waveStrips = T * (T >> 2);                // 16x64 strip per wave
    metricloss_simtile_kernel<<<(waveStrips + 7) / 8, 256, 0, stream>>>(
        Xh, label, rowsum, T);

    metricloss_finalize_kernel<<<(N + 255) / 256, 256, 0, stream>>>(
        label, rowsum, out, N);
}
